// HEALPixTransformer_609885356098
// MI455X (gfx1250) — matrix operand order repaired and checked by reading, not compiled, not run on hardware
//
#include <hip/hip_runtime.h>
#include <hip/hip_bf16.h>

typedef __attribute__((ext_vector_type(16))) _Float16 v16h;
typedef __attribute__((ext_vector_type(8)))  float    v8f;

// ---------------- problem constants (fixed by the reference) ----------------
#define BB   2
#define NTOT 3072
#define CC   4
#define PP   4
#define EE   512
#define HH   8
#define TT   256
#define FF   64
#define LL   4
#define NT   768          // NTOT / PP
#define HD   64           // EE / HH
#define E3   1536
#define E4   2048

// ---------------- generic batched/strided WMMA GEMM -------------------------
// C[b,h] = act( A[b,h] (MxK) * B[b,h] (KxN) * scale + biasv + addmat[b,h] )
// A row-major via lda; B element (k,n) at B + k*ldbk + n*ldbn; C row-major ldc.
// All M % 128 == 0, N % 64 == 0, K % 64 == 0 for every GEMM in this model.
#define BM 128
#define BN 64
#define BK 64
#define LDS_A 72   // padded f16 stride (avoids bank conflicts, keeps 16B align)
#define LDS_B 72

__launch_bounds__(256, 2)
__global__ void gemm_wmma_f16(const float* __restrict__ A, long lda, long sAb, long sAh,
                              const float* __restrict__ B, long ldbk, long ldbn, long sBb, long sBh,
                              float* __restrict__ C, long ldc, long sCb, long sCh,
                              const float* __restrict__ biasv,
                              const float* __restrict__ addmat, long am_rs, long am_cs,
                              long sMb, long sMh,
                              float scale, int act, int Hb, int K)
{
    __shared__ _Float16 As[BM * LDS_A];
    __shared__ _Float16 Bs[BN * LDS_B];

    const int bh = blockIdx.z;
    const int b  = bh / Hb;
    const int h  = bh % Hb;
    A += (long)b * sAb + (long)h * sAh;
    B += (long)b * sBb + (long)h * sBh;
    C += (long)b * sCb + (long)h * sCh;
    if (addmat) addmat += (long)b * sMb + (long)h * sMh;

    const int tid  = threadIdx.x;
    const int lane = tid & 31;
    const int wave = tid >> 5;
    const int wm   = (wave >> 1) * 32;   // wave row within 128
    const int wn   = (wave & 1) * 32;    // wave col within 64
    const int m0   = blockIdx.y * BM;
    const int n0   = blockIdx.x * BN;

    v8f acc[2][2] = {};

    // staging assignments (BK = 64)
    const int ar = tid >> 1;          // 0..127 : A row
    const int ac = (tid & 1) * 32;    // 0 or 32 : A col base (32 contiguous)
    const int bn = tid >> 2;          // 0..63  : B col (n)
    const int bk = (tid & 3) * 16;    // 0,16,32,48 : B row (k) base (16 strided)

    const int l15 = lane & 15;
    const int kbA = (lane >> 4) ? 8 : 0;    // A frag K base (ISA 16-bit A layout)
    const int kbB = (lane >> 4) ? 8 : 0;    // B frag K base (same layout as A)

    for (int k0 = 0; k0 < K; k0 += BK) {
        // stage A tile (128x64) as f16
        const float* ga = A + (long)(m0 + ar) * lda + (k0 + ac);
#pragma unroll
        for (int j = 0; j < 32; ++j)
            As[ar * LDS_A + ac + j] = (_Float16)ga[j];
        // stage B tile transposed: Bs[n][k]
        const float* gb = B + (long)(k0 + bk) * ldbk + (long)(n0 + bn) * ldbn;
#pragma unroll
        for (int j = 0; j < 16; ++j)
            Bs[bn * LDS_B + bk + j] = (_Float16)gb[(long)j * ldbk];

        // prefetch next K-slab while this one is consumed (global_prefetch_b8)
        if (k0 + BK < K) {
            __builtin_prefetch(ga + BK, 0, 3);
            __builtin_prefetch(gb + (long)BK * ldbk, 0, 3);
        }
        __syncthreads();

#pragma unroll
        for (int kk = 0; kk < BK; kk += 32) {
            v16h af[2], bf[2];
#pragma unroll
            for (int mi = 0; mi < 2; ++mi) {
                const _Float16* p = &As[(wm + mi * 16 + l15) * LDS_A + kk + kbA];
#pragma unroll
                for (int q = 0; q < 8; ++q) af[mi][q] = p[q];          // K = kbA..kbA+7
#pragma unroll
                for (int q = 0; q < 8; ++q) af[mi][8 + q] = p[16 + q]; // K = kbA+16..+23
            }
#pragma unroll
            for (int ni = 0; ni < 2; ++ni) {
                const _Float16* p = &Bs[(wn + ni * 16 + l15) * LDS_B + kk + kbB];
#pragma unroll
                for (int q = 0; q < 8; ++q) { bf[ni][q] = p[q]; bf[ni][8 + q] = p[16 + q]; } // K = kbB..+7, kbB+16..+23
            }
#pragma unroll
            for (int mi = 0; mi < 2; ++mi)
#pragma unroll
                for (int ni = 0; ni < 2; ++ni)
                    acc[mi][ni] = __builtin_amdgcn_wmma_f32_16x16x32_f16(
                        false, af[mi], false, bf[ni], (short)0, acc[mi][ni], false, false);
        }
        __syncthreads();
    }

    // epilogue: C/D layout — VGPR r, lane<16 -> M=r, lane>=16 -> M=r+8; N = lane&15
    const int mbase = m0 + wm + ((lane >> 4) ? 8 : 0);
    const int nbase = n0 + wn + l15;
#pragma unroll
    for (int mi = 0; mi < 2; ++mi) {
#pragma unroll
        for (int ni = 0; ni < 2; ++ni) {
            const int gn = nbase + ni * 16;
            const float bv = biasv ? biasv[gn] : 0.0f;
#pragma unroll
            for (int r = 0; r < 8; ++r) {
                const int gm = mbase + mi * 16 + r;
                float c = acc[mi][ni][r] * scale + bv;
                if (addmat) c += addmat[(long)gm * am_rs + (long)gn * am_cs];
                if (act == 1) { // tanh-approx GELU (jax.nn.gelu default)
                    float x = c;
                    c = 0.5f * x * (1.0f + tanhf(0.7978845608028654f * (x + 0.044715f * x * x * x)));
                }
                C[(long)gm * ldc + gn] = c;
            }
        }
    }
}

// ---------------- v = normalize(mean of 4 vec rows) -------------------------
__global__ void vnorm_kernel(const float* __restrict__ vm, float* __restrict__ v, long total)
{
    long i = (long)blockIdx.x * 256 + threadIdx.x;
    if (i >= total) return;
    const float* p = vm + i * 12;   // (B, Nt, 4, 3)
    float a = 0.f, b = 0.f, c = 0.f;
#pragma unroll
    for (int j = 0; j < 4; ++j) { a += p[j*3]; b += p[j*3+1]; c += p[j*3+2]; }
    a *= 0.25f; b *= 0.25f; c *= 0.25f;
    float inv = rsqrtf(a*a + b*b + c*c);
    v[i*3]   = a * inv;
    v[i*3+1] = b * inv;
    v[i*3+2] = c * inv;
}

// ---------------- relative-position bias (B,Nt,Nt,H) ------------------------
__global__ void bias_kernel(const float* __restrict__ v, const float* __restrict__ Wrb,
                            const float* __restrict__ brb, float* __restrict__ bias)
{
    __shared__ float Ws[FF * HH];
    __shared__ float bs[HH];
    const int tid = threadIdx.y * 16 + threadIdx.x;
    for (int i = tid; i < FF * HH; i += 256) Ws[i] = Wrb[i];
    if (tid < HH) bs[tid] = brb[tid];
    __syncthreads();

    const int b = blockIdx.z;
    const int q = blockIdx.y * 16 + threadIdx.y;
    const int k = blockIdx.x * 16 + threadIdx.x;
    const float* vq = v + ((long)b * NT + q) * 3;
    const float* vk = v + ((long)b * NT + k) * 3;
    float dx = vq[0]-vk[0], dy = vq[1]-vk[1], dz = vq[2]-vk[2];
    float dot = 1.0f - 0.5f * (dx*dx + dy*dy + dz*dz);
    dot = fminf(1.0f, fmaxf(-1.0f, dot));
    float angle = acosf(dot) * 200.0f;   // / 0.005

    float acc[HH];
#pragma unroll
    for (int h = 0; h < HH; ++h) acc[h] = bs[h];
    for (int f = 0; f < FF / 2; ++f) {
        float fr = ((float)f * (1.0f / 31.0f)) * 10.0f;   // linspace(0,1,32)/PERIOD
        float ph = fr * angle;
        float s = sinf(ph), c = cosf(ph);
#pragma unroll
        for (int h = 0; h < HH; ++h)
            acc[h] += s * Ws[f * HH + h] + c * Ws[(32 + f) * HH + h];
    }
    float* out = bias + (((long)b * NT + q) * NT + k) * HH;
#pragma unroll
    for (int h = 0; h < HH; ++h) out[h] = acc[h];
}

// ---------------- embed: h0 = xp @ W_embed + b + pos_emb --------------------
__global__ void embed_kernel(const float* __restrict__ x, const float* __restrict__ W,
                             const float* __restrict__ bvec, const float* __restrict__ pos,
                             float* __restrict__ out, long total)
{
    long i = (long)blockIdx.x * 256 + threadIdx.x;
    if (i >= total) return;
    const int e = (int)(i % EE);
    const long row = i / EE;            // b*Nt + n
    const int  n   = (int)(row % NT);
    const float* xr = x + row * (PP * CC);
    float acc = bvec[e] + pos[(long)n * EE + e];
#pragma unroll
    for (int j = 0; j < PP * CC; ++j) acc += xr[j] * W[j * EE + e];
    out[i] = acc;
}

// ---------------- adaLN: m = silu(t@W1+b1)@W2+b2 -> (B, 3E) -----------------
__global__ void adaln_kernel(const float* __restrict__ t, const float* __restrict__ W1,
                             const float* __restrict__ b1, const float* __restrict__ W2,
                             const float* __restrict__ b2, float* __restrict__ mod)
{
    __shared__ float ts[TT];
    __shared__ float hs[TT];
    const int b = blockIdx.x, tid = threadIdx.x;
    ts[tid] = t[(long)b * TT + tid];
    __syncthreads();
    float acc = b1[tid];
    for (int j = 0; j < TT; ++j) acc += ts[j] * W1[(long)j * TT + tid];
    hs[tid] = acc / (1.0f + expf(-acc));   // silu
    __syncthreads();
    for (int o = tid; o < E3; o += 256) {
        float a2 = b2[o];
        for (int j = 0; j < TT; ++j) a2 += hs[j] * W2[(long)j * E3 + o];
        mod[(long)b * E3 + o] = a2;
    }
}

// ---------------- y = (g+1)*LN(xc) + be -------------------------------------
__global__ void modln_kernel(const float* __restrict__ xc, const float* __restrict__ mod,
                             float* __restrict__ y)
{
    const long row = blockIdx.x;             // b*Nt + n
    const int  b   = (int)(row / NT);
    const float* xr = xc + row * EE;
    float* yr = y + row * EE;
    const float* g  = mod + (long)b * 3 * EE;
    const float* be = g + EE;
    const int tid = threadIdx.x;
    __shared__ float red[256];
    float s = 0.f;
    for (int i = tid; i < EE; i += 256) s += xr[i];
    red[tid] = s; __syncthreads();
    for (int st = 128; st > 0; st >>= 1) { if (tid < st) red[tid] += red[tid + st]; __syncthreads(); }
    const float mu = red[0] / EE; __syncthreads();
    float v2 = 0.f;
    for (int i = tid; i < EE; i += 256) { float d = xr[i] - mu; v2 += d * d; }
    red[tid] = v2; __syncthreads();
    for (int st = 128; st > 0; st >>= 1) { if (tid < st) red[tid] += red[tid + st]; __syncthreads(); }
    const float rstd = rsqrtf(red[0] / EE + 1e-6f);
    for (int i = tid; i < EE; i += 256)
        yr[i] = (g[i] + 1.0f) * (xr[i] - mu) * rstd + be[i];
}

// ---------------- row softmax (over last axis) ------------------------------
__global__ void softmax_rows(float* __restrict__ X, int Lrow)
{
    float* p = X + (long)blockIdx.x * Lrow;
    const int tid = threadIdx.x;
    __shared__ float red[256];
    float m = -3.0e38f;
    for (int i = tid; i < Lrow; i += 256) m = fmaxf(m, p[i]);
    red[tid] = m; __syncthreads();
    for (int s = 128; s > 0; s >>= 1) { if (tid < s) red[tid] = fmaxf(red[tid], red[tid + s]); __syncthreads(); }
    m = red[0]; __syncthreads();
    float sum = 0.f;
    for (int i = tid; i < Lrow; i += 256) { float e = expf(p[i] - m); p[i] = e; sum += e; }
    red[tid] = sum; __syncthreads();
    for (int s = 128; s > 0; s >>= 1) { if (tid < s) red[tid] += red[tid + s]; __syncthreads(); }
    const float inv = 1.0f / red[0];
    for (int i = tid; i < Lrow; i += 256) p[i] *= inv;
}

// ---------------- residual: xc += al * y / sqrt(1 + al^2) -------------------
__global__ void residual_kernel(float* __restrict__ xc, const float* __restrict__ y,
                                const float* __restrict__ mod, long total)
{
    long i = (long)blockIdx.x * 256 + threadIdx.x;
    if (i >= total) return;
    const int  e = (int)(i % EE);
    const long row = i / EE;
    const int  b = (int)(row / NT);
    const float al = mod[(long)b * 3 * EE + 2 * EE + e];
    xc[i] += al * y[i] * rsqrtf(1.0f + al * al);
}

// ---------------- host-side launcher ----------------------------------------
static void launch_gemm(hipStream_t s,
                        const float* A, long lda, long sAb, long sAh,
                        const float* B, long ldbk, long ldbn, long sBb, long sBh,
                        float* C, long ldc, long sCb, long sCh,
                        const float* biasv,
                        const float* addmat, long amrs, long amcs, long sMb, long sMh,
                        float scale, int act, int Hb,
                        int M, int N, int K, int batches)
{
    dim3 grid(N / BN, M / BM, batches);
    gemm_wmma_f16<<<grid, dim3(256), 0, s>>>(A, lda, sAb, sAh,
                                             B, ldbk, ldbn, sBb, sBh,
                                             C, ldc, sCb, sCh,
                                             biasv, addmat, amrs, amcs, sMb, sMh,
                                             scale, act, Hb, K);
}

extern "C" void kernel_launch(void* const* d_in, const int* in_sizes, int n_in,
                              void* d_out, int out_size, void* d_ws, size_t ws_size,
                              hipStream_t stream)
{
    const float* x        = (const float*)d_in[0];
    const float* t        = (const float*)d_in[1];
    const float* vec_map  = (const float*)d_in[2];
    const float* W_rb     = (const float*)d_in[3];
    const float* b_rb     = (const float*)d_in[4];
    const float* W_embed  = (const float*)d_in[5];
    const float* b_embed  = (const float*)d_in[6];
    const float* pos_emb  = (const float*)d_in[7];
    const float* ada1_W1  = (const float*)d_in[8];
    const float* ada1_b1  = (const float*)d_in[9];
    const float* ada1_W2  = (const float*)d_in[10];
    const float* ada1_b2  = (const float*)d_in[11];
    const float* qkv_W    = (const float*)d_in[12];
    const float* qkv_b    = (const float*)d_in[13];
    const float* out_W    = (const float*)d_in[14];
    const float* out_b    = (const float*)d_in[15];
    const float* ada2_W1  = (const float*)d_in[16];
    const float* ada2_b1  = (const float*)d_in[17];
    const float* ada2_W2  = (const float*)d_in[18];
    const float* ada2_b2  = (const float*)d_in[19];
    const float* mlp_W1   = (const float*)d_in[20];
    const float* mlp_b1   = (const float*)d_in[21];
    const float* mlp_W2   = (const float*)d_in[22];
    const float* mlp_b2   = (const float*)d_in[23];

    float* xc = (float*)d_out;                       // (B, Nt, E), updated in place

    // workspace layout (floats)
    float* w = (float*)d_ws;
    size_t o = 0;
    float* v_buf    = w + o; o += (size_t)BB * NT * 3;
    float* bias_buf = w + o; o += (size_t)BB * NT * NT * HH;
    float* mod1     = w + o; o += (size_t)BB * E3;
    float* mod2     = w + o; o += (size_t)BB * E3;
    float* ybuf     = w + o; o += (size_t)BB * NT * EE;
    float* qkvbuf   = w + o; o += (size_t)BB * NT * E3;
    float* logits   = w + o; o += (size_t)BB * HH * NT * NT;
    float* zbuf     = w + o; o += (size_t)BB * NT * E4;   // out-proj out + mlp1 out
    float* y2buf    = w + o; o += (size_t)BB * NT * EE;   // attn out / mlp2 out

    const long totXC = (long)BB * NT * EE;
    const float inv_sqrt_hd = 0.125f;   // 1/sqrt(64)

    // 1) v = normalize(mean patch vectors)
    vnorm_kernel<<<(BB * NT + 255) / 256, 256, 0, stream>>>(vec_map, v_buf, (long)BB * NT);
    // 2) relative position bias
    bias_kernel<<<dim3(NT / 16, NT / 16, BB), dim3(16, 16), 0, stream>>>(v_buf, W_rb, b_rb, bias_buf);
    // 3) embed
    embed_kernel<<<(int)((totXC + 255) / 256), 256, 0, stream>>>(x, W_embed, b_embed, pos_emb, xc, totXC);

    for (int l = 0; l < LL; ++l) {
        // adaLN modulation params (both branches)
        adaln_kernel<<<BB, TT, 0, stream>>>(t, ada1_W1 + (size_t)l * TT * TT, ada1_b1 + (size_t)l * TT,
                                            ada1_W2 + (size_t)l * TT * E3, ada1_b2 + (size_t)l * E3, mod1);
        adaln_kernel<<<BB, TT, 0, stream>>>(t, ada2_W1 + (size_t)l * TT * TT, ada2_b1 + (size_t)l * TT,
                                            ada2_W2 + (size_t)l * TT * E3, ada2_b2 + (size_t)l * E3, mod2);

        // y = (g1+1)*LN(xc)+be1
        modln_kernel<<<BB * NT, 256, 0, stream>>>(xc, mod1, ybuf);

        // qkv = y @ qkv_W[l] + qkv_b[l]   (M=1536, K=512, N=1536)
        launch_gemm(stream, ybuf, EE, 0, 0,
                    qkv_W + (size_t)l * EE * E3, E3, 1, 0, 0,
                    qkvbuf, E3, 0, 0,
                    qkv_b + (size_t)l * E3, nullptr, 0, 0, 0, 0,
                    1.0f, 0, 1, BB * NT, E3, EE, 1);

        // scores = q·kT * 1/sqrt(hd) + bias   per (b,h): M=N=768, K=64
        launch_gemm(stream,
                    qkvbuf, E3, (long)NT * E3, 3 * HD,                 // A = q
                    qkvbuf + HD, 1, E3, (long)NT * E3, 3 * HD,         // B(k,n) = K[n][k]
                    logits, NT, (long)HH * NT * NT, (long)NT * NT,
                    nullptr,
                    bias_buf, (long)NT * HH, HH, (long)NT * NT * HH, 1,
                    inv_sqrt_hd, 0, HH, NT, NT, HD, BB * HH);

        // softmax over k
        softmax_rows<<<BB * HH * NT, 256, 0, stream>>>(logits, NT);

        // attn @ V   per (b,h): M=768, N=64, K=768
        launch_gemm(stream,
                    logits, NT, (long)HH * NT * NT, (long)NT * NT,     // A = attn
                    qkvbuf + 2 * HD, E3, 1, (long)NT * E3, 3 * HD,     // B(k,n) = V[k][n]
                    y2buf, EE, (long)NT * EE, HD,
                    nullptr, nullptr, 0, 0, 0, 0,
                    1.0f, 0, HH, NT, HD, NT, BB * HH);

        // out-proj: z = yattn @ out_W[l] + out_b[l]
        launch_gemm(stream, y2buf, EE, 0, 0,
                    out_W + (size_t)l * EE * EE, EE, 1, 0, 0,
                    zbuf, EE, 0, 0,
                    out_b + (size_t)l * EE, nullptr, 0, 0, 0, 0,
                    1.0f, 0, 1, BB * NT, EE, EE, 1);

        // xc += al1 * z / sqrt(1+al1^2)
        residual_kernel<<<(int)((totXC + 255) / 256), 256, 0, stream>>>(xc, zbuf, mod1, totXC);

        // y = (g2+1)*LN(xc)+be2
        modln_kernel<<<BB * NT, 256, 0, stream>>>(xc, mod2, ybuf);

        // mlp1: z = gelu(y @ mlp_W1[l] + mlp_b1[l])   (N=2048)
        launch_gemm(stream, ybuf, EE, 0, 0,
                    mlp_W1 + (size_t)l * EE * E4, E4, 1, 0, 0,
                    zbuf, E4, 0, 0,
                    mlp_b1 + (size_t)l * E4, nullptr, 0, 0, 0, 0,
                    1.0f, 1, 1, BB * NT, E4, EE, 1);

        // mlp2: y2 = z @ mlp_W2[l] + mlp_b2[l]   (K=2048)
        launch_gemm(stream, zbuf, E4, 0, 0,
                    mlp_W2 + (size_t)l * E4 * EE, EE, 1, 0, 0,
                    y2buf, EE, 0, 0,
                    mlp_b2 + (size_t)l * EE, nullptr, 0, 0, 0, 0,
                    1.0f, 0, 1, BB * NT, EE, E4, 1);

        // xc += al2 * y2 / sqrt(1+al2^2)
        residual_kernel<<<(int)((totXC + 255) / 256), 256, 0, stream>>>(xc, y2buf, mod2, totXC);
    }
}
